// SALAD_full_37391985279011
// MI455X (gfx1250) — matrix operand order repaired and checked by reading, not compiled, not run on hardware
//
#include <hip/hip_runtime.h>
#include <hip/hip_bf16.h>
#include <math.h>

// ---------------------------------------------------------------------------
// SALAD head for MI455X (gfx1250, wave32, WMMA).
//   B=64, C=768, N=1024 (HxW), HID=512, L=128, M=64, G=256
// Pipeline:
//   bf16 convert -> WMMA GEMM (hidden, relu) -> WMMA GEMM (proj)
//   -> Sinkhorn (LDS, per-batch workgroup) -> WMMA agg GEMM
//   -> token MLP -> normalize/concat/normalize
// GEMM structure: 8 waves/block share one 16-column X panel; K is chunked by
// 128 with double-buffered LDS so each __syncthreads covers 4 WMMAs/wave.
// ---------------------------------------------------------------------------

typedef __attribute__((ext_vector_type(16))) __bf16 v16bf;
typedef __attribute__((ext_vector_type(8)))  __bf16 v8bf;
typedef __attribute__((ext_vector_type(8)))  float  v8f;

#define NB   64
#define NC   768
#define NN   1024
#define NCOL (NB * NN)        // 65536 total columns
#define HID  512
#define LCL  128
#define MCL  64
#define GTOK 256
#define KCH  128              // K chunk per LDS stage (4 WMMA k-steps)

// ---------------------------------------------------------------- convert
__global__ void salad_f32_to_bf16(const float* __restrict__ src,
                                  __bf16* __restrict__ dst, size_t n) {
  size_t i = (size_t)blockIdx.x * blockDim.x + threadIdx.x;
  size_t stride = (size_t)gridDim.x * blockDim.x;
  for (; i < n; i += stride) dst[i] = (__bf16)src[i];
}

// ------------------------------------------------------- WMMA GEMM + bias
// OUT[b][o][n] = act( sum_k W[o][k] * X[b][k][n] + bias[o] )
// Block: WM waves. Wave w owns the 16x16 tile (otile = blockIdx.y*WM + w).
// All waves share a 16-column x 128-K panel of X, double-buffered in LDS.
template <int O, int K, int WM, bool RELU, bool OUT_F32>
__global__ __launch_bounds__(32 * WM)
void salad_wmma_gemm(const __bf16* __restrict__ W,     // [O][K] row-major
                     const __bf16* __restrict__ X,     // [B][K][1024]
                     const float*  __restrict__ bias,  // [O]
                     void* __restrict__ out) {         // [B][O][1024]
  const int tid  = threadIdx.x;
  const int wave = tid >> 5;
  const int lane = tid & 31;
  const int hf   = lane >> 4;   // half-wave selector
  const int lm   = lane & 15;

  const int j0 = blockIdx.x * 16;           // global column of 16-wide tile
  const int b  = j0 >> 10;                  // batch (1024 cols per batch)
  const int n0 = j0 & 1023;
  const int otile = (blockIdx.y * WM + wave) * 16;

  // double-buffered transposed panel: sX[buf][n][k], 4KB per buffer
  __shared__ alignas(32) __bf16 sX[2][16 * KCH];

  // fold bias into the accumulator (D layout: VGPR r -> M = r + hf*8)
  v8f acc;
  for (int r = 0; r < 8; ++r) acc[r] = bias[otile + r + hf * 8];

  const __bf16* Xb   = X + (size_t)b * K * NN + n0;
  const __bf16* Wrow = W + (size_t)(otile + lm) * K;

  constexpr int NCHUNK = K / KCH;
  constexpr int T = 32 * WM;

  // stage chunk 0
  for (int t = tid; t < 16 * KCH; t += T) {
    const int k = t >> 4;
    const int n = t & 15;
    sX[0][n * KCH + k] = Xb[(size_t)k * NN + n];
  }
  __syncthreads();

  for (int c = 0; c < NCHUNK; ++c) {
    // stage next chunk into the other buffer (overlaps current compute)
    if (c + 1 < NCHUNK) {
      const int kb0 = (c + 1) * KCH;
      __bf16* dstb = sX[(c + 1) & 1];
      for (int t = tid; t < 16 * KCH; t += T) {
        const int k = t >> 4;
        const int n = t & 15;
        dstb[n * KCH + k] = Xb[(size_t)(kb0 + k) * NN + n];
      }
      // pull the next weight panel toward the caches (global_prefetch_b8)
      __builtin_prefetch(Wrow + kb0, 0, 0);
    }

    const __bf16* curb = sX[c & 1];
    #pragma unroll
    for (int kk = 0; kk < KCH / 32; ++kk) {
      // A fragment (16x32 bf16): lane(m=lm) holds K runs
      //   [kb2, kb2+8) and [kb2+16, kb2+24) -> two b128 global loads
      v16bf a;
      {
        const int kb2 = c * KCH + kk * 32 + hf * 8;
        const v8bf r1 = *(const v8bf*)(Wrow + kb2);
        const v8bf r2 = *(const v8bf*)(Wrow + kb2 + 16);
        for (int i = 0; i < 8; ++i) { a[i] = r1[i]; a[i + 8] = r2[i]; }
      }
      // B fragment (32x16 bf16): lane(col=lm) holds K runs [hf*8, +8) and
      // [16+hf*8, +8) in the transposed LDS panel -> two ds_load_b128
      v16bf bf;
      {
        const __bf16* src = curb + lm * KCH + kk * 32 + hf * 8;
        const v8bf b1 = *(const v8bf*)(src);
        const v8bf b2 = *(const v8bf*)(src + 16);
        for (int i = 0; i < 8; ++i) { bf[i] = b1[i]; bf[i + 8] = b2[i]; }
      }
      acc = __builtin_amdgcn_wmma_f32_16x16x32_bf16(
          false, a, false, bf, (short)0, acc, false, false);
    }
    __syncthreads();   // writes of next buffer done; reads of current done
  }

  const int ncol = n0 + lm;
  for (int r = 0; r < 8; ++r) {
    float vv = acc[r];
    if (RELU) vv = fmaxf(vv, 0.0f);
    const size_t idx = ((size_t)b * O + otile + r + hf * 8) * NN + ncol;
    if (OUT_F32) ((float*)out)[idx]  = vv;
    else         ((__bf16*)out)[idx] = (__bf16)vv;
  }
}

// ------------------------------------------------------------- Sinkhorn
// One workgroup (256 threads) per batch. u[65], v[1024] live in LDS;
// Z rows are re-read from the f32 score buffer (L2-resident, 16 MB total).
__global__ __launch_bounds__(256)
void salad_sinkhorn(const float* __restrict__ p,     // [B][64][1024]
                    const float* __restrict__ dust,  // scalar alpha
                    __bf16* __restrict__ P) {        // [B][64][1024]
  const int b   = blockIdx.x;
  const int tid = threadIdx.x;
  const int wave = tid >> 5;
  const int lane = tid & 31;

  __shared__ float u[65];
  __shared__ float v[1024];

  const float alpha   = dust[0];
  const float norm    = -__logf(1088.0f);        // -log(m+n), m=64 n=1024
  const float lmu_bin = __logf(960.0f) + norm;   // log(n-m) + norm
  const float* pb = p + (size_t)b * MCL * NN;

  for (int i = tid; i < NN; i += 256) v[i] = 0.0f;
  if (tid < 65) u[tid] = 0.0f;
  __syncthreads();

  for (int it = 0; it < 3; ++it) {
    // u[m] = log_mu[m] - logsumexp_n(Z[m][n] + v[n]); one wave per row slice
    for (int m = wave; m < 65; m += 8) {
      float mx = -1e30f, s = 0.0f;
      for (int n = lane; n < NN; n += 32) {
        const float z = ((m < MCL) ? pb[m * NN + n] : alpha) + v[n];
        const float nm = fmaxf(mx, z);
        s = s * __expf(mx - nm) + __expf(z - nm);
        mx = nm;
      }
      for (int off = 16; off > 0; off >>= 1) {
        const float omx = __shfl_xor(mx, off, 32);
        const float os  = __shfl_xor(s,  off, 32);
        const float nm  = fmaxf(mx, omx);
        s = s * __expf(mx - nm) + os * __expf(omx - nm);
        mx = nm;
      }
      if (lane == 0)
        u[m] = ((m < MCL) ? norm : lmu_bin) - (mx + __logf(s));
    }
    __syncthreads();
    // v[n] = log_nu - logsumexp_m(Z[m][n] + u[m]); coalesced column sweep
    for (int n = tid; n < NN; n += 256) {
      float mx = -1e30f, s = 0.0f;
      for (int m = 0; m < 65; ++m) {
        const float z = ((m < MCL) ? pb[m * NN + n] : alpha) + u[m];
        const float nm = fmaxf(mx, z);
        s = s * __expf(mx - nm) + __expf(z - nm);
        mx = nm;
      }
      v[n] = norm - (mx + __logf(s));
    }
    __syncthreads();
  }

  // P = exp(Z + u + v - norm), dust-bin row dropped
  __bf16* Pb = P + (size_t)b * MCL * NN;
  for (int i = tid; i < MCL * NN; i += 256) {
    const int m = i >> 10, n = i & 1023;
    Pb[i] = (__bf16)__expf(pb[i] + u[m] + v[n] - norm);
  }
}

// ------------------------------------------- aggregation GEMM (f x P^T)
// agg[b][l][m] = sum_n F[b][l][n] * P[b][m][n].  Both operands row-major
// over K=n, so A and B fragments are contiguous b128 global loads.
__global__ __launch_bounds__(32)
void salad_wmma_agg(const __bf16* __restrict__ F,   // [B][128][1024]
                    const __bf16* __restrict__ P,   // [B][64][1024]
                    float* __restrict__ agg) {      // [B][128][64]
  const int lane = threadIdx.x & 31;
  const int hf = lane >> 4, lm = lane & 15;
  const int mtile = blockIdx.x * 16;   // 0..63
  const int ltile = blockIdx.y * 16;   // 0..127
  const int b     = blockIdx.z;

  const __bf16* Frow = F + ((size_t)b * LCL + ltile + lm) * NN;
  const __bf16* Prow = P + ((size_t)b * MCL + mtile + lm) * NN;

  v8f acc;
  for (int r = 0; r < 8; ++r) acc[r] = 0.0f;

  for (int kb = 0; kb < NN; kb += 32) {
    v16bf a, bf;
    const int kb2 = kb + hf * 8;
    const v8bf r1 = *(const v8bf*)(Frow + kb2);
    const v8bf r2 = *(const v8bf*)(Frow + kb2 + 16);
    for (int i = 0; i < 8; ++i) { a[i] = r1[i]; a[i + 8] = r2[i]; }
    const int ko = kb + hf * 8;
    const v8bf b1 = *(const v8bf*)(Prow + ko);
    const v8bf b2 = *(const v8bf*)(Prow + ko + 16);
    for (int i = 0; i < 8; ++i) { bf[i] = b1[i]; bf[i + 8] = b2[i]; }
    acc = __builtin_amdgcn_wmma_f32_16x16x32_bf16(
        false, a, false, bf, (short)0, acc, false, false);
  }
  for (int r = 0; r < 8; ++r)
    agg[((size_t)b * LCL + ltile + r + hf * 8) * MCL + mtile + lm] = acc[r];
}

// ----------------------------------------------------------- token MLP
__global__ void salad_token_mlp(const float* __restrict__ in,   // [B][C]
                                const float* __restrict__ w,    // [O][C]
                                const float* __restrict__ bias, // [O]
                                float* __restrict__ out,        // [B][O]
                                int O, int C, int relu) {
  const int idx = blockIdx.x * blockDim.x + threadIdx.x;
  if (idx >= NB * O) return;
  const int b = idx / O, o = idx % O;
  const float* ir = in + (size_t)b * C;
  const float* wr = w  + (size_t)o * C;
  float s = bias[o];
  for (int c = 0; c < C; ++c) s += ir[c] * wr[c];
  out[idx] = relu ? fmaxf(s, 0.0f) : s;
}

// ------------------------------------------------- normalize + concat
__global__ __launch_bounds__(256)
void salad_finalize(const float* __restrict__ tok,   // [B][256]
                    const float* __restrict__ agg,   // [B][128][64]
                    float* __restrict__ out) {       // [B][8448]
  const int b = blockIdx.x, tid = threadIdx.x;
  __shared__ float cn[MCL];
  __shared__ float tokn[GTOK];
  __shared__ float red[256];
  const float* aggb = agg + (size_t)b * LCL * MCL;

  // token L2 norm
  const float tv = tok[(size_t)b * GTOK + tid];
  red[tid] = tv * tv;
  __syncthreads();
  for (int s = 128; s > 0; s >>= 1) {
    if (tid < s) red[tid] += red[tid + s];
    __syncthreads();
  }
  const float tn = fmaxf(sqrtf(red[0]), 1e-12f);

  // per-cluster column norms over l (intra-normalization)
  if (tid < MCL) {
    float s = 0.0f;
    for (int l = 0; l < LCL; ++l) {
      const float a = aggb[l * MCL + tid];
      s += a * a;
    }
    cn[tid] = fmaxf(sqrtf(s), 1e-12f);
  }
  __syncthreads();       // cn ready; also fences red[0] reads before reuse
  tokn[tid] = tv / tn;

  // global sum of squares of the concatenated vector
  float part = (tv / tn) * (tv / tn);
  for (int i = tid; i < LCL * MCL; i += 256) {
    const float a = aggb[i] / cn[i & 63];
    part += a * a;
  }
  red[tid] = part;
  __syncthreads();
  for (int s = 128; s > 0; s >>= 1) {
    if (tid < s) red[tid] += red[tid + s];
    __syncthreads();
  }
  const float inv = 1.0f / fmaxf(sqrtf(red[0]), 1e-12f);

  float* ob = out + (size_t)b * (GTOK + LCL * MCL);
  ob[tid] = tokn[tid] * inv;
  for (int i = tid; i < LCL * MCL; i += 256)
    ob[GTOK + i] = (aggb[i] / cn[i & 63]) * inv;
}

// ---------------------------------------------------------------------------
extern "C" void kernel_launch(void* const* d_in, const int* in_sizes, int n_in,
                              void* d_out, int out_size, void* d_ws, size_t ws_size,
                              hipStream_t stream) {
  const float* x    = (const float*)d_in[0];
  const float* t    = (const float*)d_in[1];
  const float* tw1  = (const float*)d_in[2];
  const float* tb1  = (const float*)d_in[3];
  const float* tw2  = (const float*)d_in[4];
  const float* tb2  = (const float*)d_in[5];
  const float* cw1  = (const float*)d_in[6];
  const float* cb1  = (const float*)d_in[7];
  const float* cw2  = (const float*)d_in[8];
  const float* cb2  = (const float*)d_in[9];
  const float* sw1  = (const float*)d_in[10];
  const float* sb1  = (const float*)d_in[11];
  const float* sw2  = (const float*)d_in[12];
  const float* sb2  = (const float*)d_in[13];
  const float* dust = (const float*)d_in[14];

  // workspace carve-out (256B aligned)
  char* wp = (char*)d_ws;
  auto carve = [&](size_t bytes) {
    char* p = wp;
    wp += (bytes + 255) & ~(size_t)255;
    return p;
  };
  __bf16* Xb  = (__bf16*)carve((size_t)NC  * NCOL * 2);   // x as bf16
  __bf16* Hb  = (__bf16*)carve((size_t)HID * NCOL * 2);   // hidden (reused)
  __bf16* Fb  = (__bf16*)carve((size_t)LCL * NCOL * 2);   // cluster feats
  float*  Pf  = (float*) carve((size_t)MCL * NCOL * 4);   // scores (f32)
  __bf16* Pb  = (__bf16*)carve((size_t)MCL * NCOL * 2);   // transport plan
  float*  agg = (float*) carve((size_t)NB * LCL * MCL * 4);
  float*  tkh = (float*) carve((size_t)NB * HID * 4);
  float*  tkv = (float*) carve((size_t)NB * GTOK * 4);
  __bf16* Wc1 = (__bf16*)carve((size_t)HID * NC * 2);
  __bf16* Ws1 = (__bf16*)carve((size_t)HID * NC * 2);
  __bf16* Wc2 = (__bf16*)carve((size_t)LCL * HID * 2);
  __bf16* Ws2 = (__bf16*)carve((size_t)MCL * HID * 2);

  // --- bf16 conversions
  salad_f32_to_bf16<<<4096, 256, 0, stream>>>(x, Xb, (size_t)NC * NCOL);
  salad_f32_to_bf16<<<512, 256, 0, stream>>>(cw1, Wc1, (size_t)HID * NC);
  salad_f32_to_bf16<<<512, 256, 0, stream>>>(sw1, Ws1, (size_t)HID * NC);
  salad_f32_to_bf16<<<64,  256, 0, stream>>>(cw2, Wc2, (size_t)LCL * HID);
  salad_f32_to_bf16<<<64,  256, 0, stream>>>(sw2, Ws2, (size_t)MCL * HID);

  // cluster path: h = relu(cw1 @ x + cb1); f = cw2 @ h + cb2
  salad_wmma_gemm<HID, NC, 8, true,  false>
      <<<dim3(NCOL / 16, HID / 128), 256, 0, stream>>>(Wc1, Xb, cb1, Hb);
  salad_wmma_gemm<LCL, HID, 8, false, false>
      <<<dim3(NCOL / 16, 1), 256, 0, stream>>>(Wc2, Hb, cb2, Fb);
  // score path: hs = relu(sw1 @ x + sb1); p = sw2 @ hs + sb2 (f32 out)
  salad_wmma_gemm<HID, NC, 8, true,  false>
      <<<dim3(NCOL / 16, HID / 128), 256, 0, stream>>>(Ws1, Xb, sb1, Hb);
  salad_wmma_gemm<MCL, HID, 4, false, true>
      <<<dim3(NCOL / 16, 1), 128, 0, stream>>>(Ws2, Hb, sb2, Pf);

  // Sinkhorn transport (one workgroup per batch)
  salad_sinkhorn<<<NB, 256, 0, stream>>>(Pf, dust, Pb);

  // VLAD aggregation: agg[b] = f[b] @ P[b]^T
  salad_wmma_agg<<<dim3(MCL / 16, LCL / 16, NB), 32, 0, stream>>>(Fb, Pb, agg);

  // token MLP (tiny)
  salad_token_mlp<<<(NB * HID + 255) / 256, 256, 0, stream>>>(t, tw1, tb1, tkh, HID, NC, 1);
  salad_token_mlp<<<(NB * GTOK + 255) / 256, 256, 0, stream>>>(tkh, tw2, tb2, tkv, GTOK, HID, 0);

  // normalize + concat + normalize
  salad_finalize<<<NB, 256, 0, stream>>>(tkv, agg, (float*)d_out);
}
